// _MambaBlock_9096740733534
// MI455X (gfx1250) — matrix operand order repaired and checked by reading, not compiled, not run on hardware
//
#include <hip/hip_runtime.h>
#include <hip/hip_bf16.h>

typedef __attribute__((ext_vector_type(16))) _Float16 v16h;
typedef __attribute__((ext_vector_type(8)))  _Float16 v8h;
typedef __attribute__((ext_vector_type(8)))  float    v8f;

#define D_MODEL 768
#define D_STATE 16
#define D_CONV  4
#define D_INNER 1536
#define DT_RANK 48
#define BB      4
#define LL      2048
#define MROWS   (BB * LL)        // 8192
#define XPROJ_N 80               // DT_RANK + 2*D_STATE
#define XPROJ_NP 128             // padded to tile multiple
#define DT_KP   64               // DT_RANK padded to 2*K-steps

static __device__ __forceinline__ float sigmoidf_(float x) {
    return 1.0f / (1.0f + __expf(-x));
}

// Async global -> LDS copy of 16 bytes per lane (CDNA5, ASYNCcnt-tracked).
// LDS address = low 32 bits of the generic pointer (flat aperture rule).
static __device__ __forceinline__ void async_copy_b128(const void* gptr, void* lptr) {
    unsigned lds = (unsigned)(size_t)lptr;
    asm volatile("global_load_async_to_lds_b128 %0, %1, off"
                 :: "v"(lds), "v"(gptr) : "memory");
}
static __device__ __forceinline__ void async_wait0() {
    asm volatile("s_wait_asynccnt 0x0" ::: "memory");
}

// ---------------------------------------------------------------------------
// Convert fp32 weight W (K x N, row-major) into f16 transposed WT (Np x Kp,
// row-major over N), zero-padded.  One-time per launch, bandwidth-trivial.
// ---------------------------------------------------------------------------
__global__ __launch_bounds__(256)
void k_convert_padT(const float* __restrict__ W, _Float16* __restrict__ WT,
                    int K, int N, int Kp, int Np) {
    int idx = blockIdx.x * 256 + threadIdx.x;
    if (idx >= Np * Kp) return;
    int n = idx / Kp;
    int k = idx % Kp;
    float v = (n < N && k < K) ? W[(size_t)k * N + n] : 0.f;
    WT[idx] = (_Float16)v;
}

// ---------------------------------------------------------------------------
// RMSNorm over D_MODEL, fp32 in -> f16 out
// ---------------------------------------------------------------------------
__global__ __launch_bounds__(256)
void k_rmsnorm(const float* __restrict__ x, const float* __restrict__ w,
               _Float16* __restrict__ xn) {
    __shared__ float red[256];
    const int row = blockIdx.x;
    const float* xr = x + (size_t)row * D_MODEL;
    float s = 0.f;
    for (int i = threadIdx.x; i < D_MODEL; i += 256) {
        float v = xr[i];
        s += v * v;
    }
    red[threadIdx.x] = s;
    __syncthreads();
    for (int off = 128; off > 0; off >>= 1) {
        if (threadIdx.x < off) red[threadIdx.x] += red[threadIdx.x + off];
        __syncthreads();
    }
    const float inv = rsqrtf(red[0] / (float)D_MODEL + 1e-5f);
    for (int i = threadIdx.x; i < D_MODEL; i += 256) {
        xn[(size_t)row * D_MODEL + i] = (_Float16)(xr[i] * inv * w[i]);
    }
}

// ---------------------------------------------------------------------------
// WMMA GEMM, guard-free: C(MxN) = A(MxK, f16) * BT^T   (BT is N-major f16)
// All of M, N, K are multiples of the tile (128, 128, 32).
// Block tile 128x128: 8 waves as 4(M) x 2(N); each wave 32x64 via 2x4 frags
// -> 8 v_wmma per 32-wide K-step per wave.
// Staging: GLOBAL_LOAD_ASYNC_TO_LDS_B128 into a ping-pong double buffer so the
// next tile's DMA overlaps the current tile's WMMA work (ASYNCcnt-tracked).
// mode 0: store f16   mode 1: softplus(acc+bias[n]) f16   mode 2: resid+acc f32
// ---------------------------------------------------------------------------
__global__ __launch_bounds__(256)
void k_gemm_wmma(const _Float16* __restrict__ A, int lda,
                 const _Float16* __restrict__ BT, int ldbt,
                 int M, int N, int K, int mode,
                 const float* __restrict__ bias,
                 const float* __restrict__ resid,
                 _Float16* __restrict__ Ch,
                 float* __restrict__ Cf, int ldc) {
    __shared__ __align__(16) _Float16 sA[2][128][40];  // row stride 80B
    __shared__ __align__(16) _Float16 sB[2][128][40];  // N-major: sB[b][n][k]

    const int tid  = threadIdx.x;
    const int lane = tid & 31;
    const int wave = tid >> 5;
    const int wm = wave >> 1;          // 0..3  (M direction, 32 rows each)
    const int wn = wave & 1;           // 0..1  (N direction, 64 cols each)
    const int lr = lane & 15;
    const int hi = lane >> 4;
    const int m0 = blockIdx.y * 128;
    const int n0 = blockIdx.x * 128;

    v8f acc[2][4] = {};

    auto stage = [&](int buf, int k0) {
        #pragma unroll
        for (int it = 0; it < 2; ++it) {
            int e = tid * 8 + it * 2048;
            int r = e >> 5, c = e & 31;
            async_copy_b128(&A[(size_t)(m0 + r) * lda + k0 + c],  &sA[buf][r][c]);
            async_copy_b128(&BT[(size_t)(n0 + r) * ldbt + k0 + c], &sB[buf][r][c]);
        }
    };

    const int nT = K >> 5;   // K / 32
    stage(0, 0);

    for (int t = 0; t < nT; ++t) {
        async_wait0();       // this wave's DMA into buffer t&1 done
        __syncthreads();     // ... and everyone's; also fences ping-pong reuse
        if (t + 1 < nT) stage((t + 1) & 1, (t + 1) << 5);

        const int buf = t & 1;

        // ---- assemble fragments per ISA 7.12.2 layouts ----
        v16h af[2], bf[4];
        #pragma unroll
        for (int mi = 0; mi < 2; ++mi) {
            int ar = wm * 32 + mi * 16 + lr;
            v8h lo = *(const v8h*)&sA[buf][ar][hi * 8];        // K = hi*8 .. +7
            v8h hv = *(const v8h*)&sA[buf][ar][16 + hi * 8];   // K = 16+hi*8 .. +7
            #pragma unroll
            for (int i = 0; i < 8; ++i) { af[mi][i] = lo[i]; af[mi][8 + i] = hv[i]; }
        }
        #pragma unroll
        for (int ni = 0; ni < 4; ++ni) {
            int bn = wn * 64 + ni * 16 + lr;
            v8h lo = *(const v8h*)&sB[buf][bn][hi * 8];        // K = hi*8 .. +7
            v8h hv = *(const v8h*)&sB[buf][bn][16 + hi * 8];   // K = 16+hi*8 .. +7
            #pragma unroll
            for (int i = 0; i < 8; ++i) { bf[ni][i] = lo[i]; bf[ni][8 + i] = hv[i]; }
        }

        #pragma unroll
        for (int mi = 0; mi < 2; ++mi)
            #pragma unroll
            for (int ni = 0; ni < 4; ++ni)
                acc[mi][ni] = __builtin_amdgcn_wmma_f32_16x16x32_f16(
                    false, af[mi], false, bf[ni], (short)0, acc[mi][ni], false, false);
    }

    // ---- epilogue: C/D layout -> VGPR r holds M = hi*8 + r, lane holds N = lr ----
    #pragma unroll
    for (int mi = 0; mi < 2; ++mi) {
        #pragma unroll
        for (int ni = 0; ni < 4; ++ni) {
            int rbase = m0 + wm * 32 + mi * 16 + hi * 8;
            int col   = n0 + wn * 64 + ni * 16 + lr;
            #pragma unroll
            for (int r = 0; r < 8; ++r) {
                int row = rbase + r;
                float v = acc[mi][ni][r];
                if (mode == 0) {
                    Ch[(size_t)row * ldc + col] = (_Float16)v;
                } else if (mode == 1) {
                    float t = v + bias[col];
                    t = (t > 20.f) ? t : log1pf(__expf(t));   // softplus
                    Ch[(size_t)row * ldc + col] = (_Float16)t;
                } else {
                    Cf[(size_t)row * ldc + col] = resid[(size_t)row * ldc + col] + v;
                }
            }
        }
    }
}

// ---------------------------------------------------------------------------
// Depthwise causal conv (width 4) + bias + SiLU.  xi = xz[:, :1536]
// ---------------------------------------------------------------------------
__global__ __launch_bounds__(256)
void k_conv_silu(const _Float16* __restrict__ xz, const float* __restrict__ cw,
                 const float* __restrict__ cb, _Float16* __restrict__ xc) {
    int idx = blockIdx.x * 256 + threadIdx.x;
    if (idx >= MROWS * D_INNER) return;
    int c   = idx % D_INNER;
    int row = idx / D_INNER;
    int l   = row % LL;
    float acc = cb[c];
    #pragma unroll
    for (int j = 0; j < D_CONV; ++j) {
        int li = l - (D_CONV - 1) + j;
        if (li >= 0) {
            float xv = (float)xz[(size_t)(row - (D_CONV - 1) + j) * (2 * D_INNER) + c];
            acc += xv * cw[c * D_CONV + j];
        }
    }
    acc = acc * sigmoidf_(acc);  // SiLU
    xc[(size_t)row * D_INNER + c] = (_Float16)acc;
}

// ---------------------------------------------------------------------------
// Selective scan: one thread per (b, d) channel, 16-state recurrence over L.
// Fuses y = y + D*x, then gate with silu(z), stores f16 for out_proj.
// ---------------------------------------------------------------------------
__global__ __launch_bounds__(256)
void k_scan(const _Float16* __restrict__ dtb, const _Float16* __restrict__ xc,
            const _Float16* __restrict__ xdbl, const _Float16* __restrict__ xz,
            const float* __restrict__ A_log, const float* __restrict__ Dp,
            _Float16* __restrict__ yin) {
    int g = blockIdx.x * 256 + threadIdx.x;
    if (g >= BB * D_INNER) return;
    int b = g / D_INNER;
    int d = g % D_INNER;

    float As[D_STATE];
    #pragma unroll
    for (int s = 0; s < D_STATE; ++s) As[s] = -__expf(A_log[d * D_STATE + s]);
    float h[D_STATE];
    #pragma unroll
    for (int s = 0; s < D_STATE; ++s) h[s] = 0.f;

    const float Dv = Dp[d];
    for (int l = 0; l < LL; ++l) {
        size_t row = (size_t)b * LL + l;
        float dt  = (float)dtb[row * D_INNER + d];
        float x   = (float)xc[row * D_INNER + d];
        float dtx = dt * x;
        const _Float16* xd = xdbl + row * XPROJ_NP;
        float y = 0.f;
        #pragma unroll
        for (int s = 0; s < D_STATE; ++s) {
            float Bv = (float)xd[DT_RANK + s];
            float Cv = (float)xd[DT_RANK + D_STATE + s];
            h[s] = h[s] * __expf(dt * As[s]) + dtx * Bv;
            y += h[s] * Cv;
        }
        y += Dv * x;
        float z = (float)xz[row * (2 * D_INNER) + D_INNER + d];
        y *= z * sigmoidf_(z);
        yin[row * D_INNER + d] = (_Float16)y;
    }
}

// ---------------------------------------------------------------------------
extern "C" void kernel_launch(void* const* d_in, const int* in_sizes, int n_in,
                              void* d_out, int out_size, void* d_ws, size_t ws_size,
                              hipStream_t stream) {
    const float* seq        = (const float*)d_in[0];
    const float* norm_w     = (const float*)d_in[1];
    const float* in_proj_w  = (const float*)d_in[2];
    const float* conv_w     = (const float*)d_in[3];
    const float* conv_b     = (const float*)d_in[4];
    const float* x_proj_w   = (const float*)d_in[5];
    const float* dt_proj_w  = (const float*)d_in[6];
    const float* dt_bias    = (const float*)d_in[7];
    const float* A_log      = (const float*)d_in[8];
    const float* D_param    = (const float*)d_in[9];
    const float* out_proj_w = (const float*)d_in[10];
    float* out = (float*)d_out;

    char* ws = (char*)d_ws;
    size_t off = 0;
    auto alloc = [&](size_t bytes) -> void* {
        void* p = ws + off;
        off = (off + bytes + 255) & ~(size_t)255;
        return p;
    };
    _Float16* xn    = (_Float16*)alloc((size_t)MROWS * D_MODEL * 2);
    _Float16* xz    = (_Float16*)alloc((size_t)MROWS * 2 * D_INNER * 2);
    _Float16* xc    = (_Float16*)alloc((size_t)MROWS * D_INNER * 2);
    _Float16* xdbl  = (_Float16*)alloc((size_t)MROWS * XPROJ_NP * 2);
    _Float16* dtb   = (_Float16*)alloc((size_t)MROWS * D_INNER * 2);
    _Float16* yin   = (_Float16*)alloc((size_t)MROWS * D_INNER * 2);
    _Float16* wT_in  = (_Float16*)alloc((size_t)(2 * D_INNER) * D_MODEL * 2);
    _Float16* wT_x   = (_Float16*)alloc((size_t)XPROJ_NP * D_INNER * 2);
    _Float16* wT_dt  = (_Float16*)alloc((size_t)D_INNER * DT_KP * 2);
    _Float16* wT_out = (_Float16*)alloc((size_t)D_MODEL * D_INNER * 2);

    auto cdiv = [](int a, int b) { return (a + b - 1) / b; };

    // 0) one-time weight convert/transpose/pad to f16 (N-major)
    k_convert_padT<<<cdiv((2 * D_INNER) * D_MODEL, 256), 256, 0, stream>>>(
        in_proj_w, wT_in, D_MODEL, 2 * D_INNER, D_MODEL, 2 * D_INNER);
    k_convert_padT<<<cdiv(XPROJ_NP * D_INNER, 256), 256, 0, stream>>>(
        x_proj_w, wT_x, D_INNER, XPROJ_N, D_INNER, XPROJ_NP);
    k_convert_padT<<<cdiv(D_INNER * DT_KP, 256), 256, 0, stream>>>(
        dt_proj_w, wT_dt, DT_RANK, D_INNER, DT_KP, D_INNER);
    k_convert_padT<<<cdiv(D_MODEL * D_INNER, 256), 256, 0, stream>>>(
        out_proj_w, wT_out, D_INNER, D_MODEL, D_INNER, D_MODEL);

    // 1) RMSNorm -> f16
    k_rmsnorm<<<MROWS, 256, 0, stream>>>(seq, norm_w, xn);

    // 2) in_proj: (8192x768) x (768x3072) -> xz f16
    {
        dim3 g((2 * D_INNER) / 128, MROWS / 128);
        k_gemm_wmma<<<g, 256, 0, stream>>>(xn, D_MODEL, wT_in, D_MODEL,
                                           MROWS, 2 * D_INNER, D_MODEL, 0,
                                           nullptr, nullptr, xz, nullptr, 2 * D_INNER);
    }

    // 3) depthwise conv + SiLU -> xc f16
    k_conv_silu<<<cdiv(MROWS * D_INNER, 256), 256, 0, stream>>>(xz, conv_w, conv_b, xc);

    // 4) x_proj: (8192x1536) x (1536x128pad) -> xdbl f16 (cols 80..127 = 0)
    {
        dim3 g(XPROJ_NP / 128, MROWS / 128);
        k_gemm_wmma<<<g, 256, 0, stream>>>(xc, D_INNER, wT_x, D_INNER,
                                           MROWS, XPROJ_NP, D_INNER, 0,
                                           nullptr, nullptr, xdbl, nullptr, XPROJ_NP);
    }

    // 5) dt_proj + bias + softplus: (8192x64pad) x (64pad x1536) -> dtb f16
    //    (weight rows 48..63 are zero, so A cols 48..63 contribute nothing)
    {
        dim3 g(D_INNER / 128, MROWS / 128);
        k_gemm_wmma<<<g, 256, 0, stream>>>(xdbl, XPROJ_NP, wT_dt, DT_KP,
                                           MROWS, D_INNER, DT_KP, 1,
                                           dt_bias, nullptr, dtb, nullptr, D_INNER);
    }

    // 6) selective scan + D*x + silu(z) gate -> yin f16
    k_scan<<<cdiv(BB * D_INNER, 256), 256, 0, stream>>>(dtb, xc, xdbl, xz,
                                                        A_log, D_param, yin);

    // 7) out_proj + residual: (8192x1536) x (1536x768) + seq -> out f32
    {
        dim3 g(D_MODEL / 128, MROWS / 128);
        k_gemm_wmma<<<g, 256, 0, stream>>>(yin, D_INNER, wT_out, D_INNER,
                                           MROWS, D_MODEL, D_INNER, 2,
                                           nullptr, seq, nullptr, out, D_MODEL);
    }
}
